// SwitchFFN_47132971106701
// MI455X (gfx1250) — matrix operand order repaired and checked by reading, not compiled, not run on hardware
//
#include <hip/hip_runtime.h>
#include <math.h>

#define T_TOK   8192
#define DD      1024
#define FF      4096
#define EE      8
#define TILE_M  64
#define TPAD    (T_TOK + TILE_M * EE)   // 8704: each expert segment 64-aligned
#define NM      (TPAD / TILE_M)         // 136 m-tiles
#define LN_EPS  1e-5f

typedef __attribute__((ext_vector_type(16))) __bf16 v16bf;
typedef __attribute__((ext_vector_type(8)))  float  v8f;

// ---------- bf16 helpers ----------
__device__ __forceinline__ unsigned short f32_to_bf16u(float f) {  // RNE
  unsigned int u = __float_as_uint(f);
  unsigned int r = u + 0x7fffu + ((u >> 16) & 1u);
  return (unsigned short)(r >> 16);
}
// pack two f32 -> two bf16 (truncate) with one v_perm_b32
__device__ __forceinline__ unsigned int pack_bf16_pair(float lo, float hi) {
  return __builtin_amdgcn_perm(__float_as_uint(hi), __float_as_uint(lo), 0x07060302u);
}

// =====================================================================
// Kernel 1: router (softmax top-1 gate/idx on raw x) + layernorm -> bf16 xn
// =====================================================================
__global__ __launch_bounds__(128) void router_ln_kernel(
    const float* __restrict__ x, const float* __restrict__ Wr, const float* __restrict__ br,
    const float* __restrict__ ln_g, const float* __restrict__ ln_b,
    unsigned short* __restrict__ xn, float* __restrict__ gate, int* __restrict__ expert) {
  int t = blockIdx.x;
  int tid = threadIdx.x;
  const float* xr = x + (size_t)t * DD;
  int dBase = tid * 8;
  float v[8];
#pragma unroll
  for (int i = 0; i < 8; ++i) v[i] = xr[dBase + i];

  float sum = 0.f, sq = 0.f;
  float lg[EE];
#pragma unroll
  for (int e = 0; e < EE; ++e) lg[e] = 0.f;
#pragma unroll
  for (int i = 0; i < 8; ++i) {
    sum += v[i];
    sq  += v[i] * v[i];
    const float* wr = Wr + (size_t)(dBase + i) * EE;
#pragma unroll
    for (int e = 0; e < EE; ++e) lg[e] += v[i] * wr[e];
  }

  __shared__ float red[10][128];
  red[0][tid] = sum;
  red[1][tid] = sq;
#pragma unroll
  for (int e = 0; e < EE; ++e) red[2 + e][tid] = lg[e];
  __syncthreads();
  for (int s = 64; s > 0; s >>= 1) {
    if (tid < s) {
#pragma unroll
      for (int j = 0; j < 10; ++j) red[j][tid] += red[j][tid + s];
    }
    __syncthreads();
  }

  __shared__ float s_mean, s_rstd;
  if (tid == 0) {
    float mean = red[0][0] * (1.0f / DD);
    float var  = red[1][0] * (1.0f / DD) - mean * mean;
    s_mean = mean;
    s_rstd = rsqrtf(var + LN_EPS);
    float l[EE];
    float lmax = -1e30f;
    int idx = 0;
#pragma unroll
    for (int e = 0; e < EE; ++e) {
      l[e] = red[2 + e][0] + br[e];
      if (l[e] > lmax) { lmax = l[e]; idx = e; }
    }
    float ssum = 0.f;
#pragma unroll
    for (int e = 0; e < EE; ++e) ssum += expf(l[e] - lmax);
    gate[t] = 1.0f / ssum;
    expert[t] = idx;
  }
  __syncthreads();
  float mean = s_mean, rstd = s_rstd;
#pragma unroll
  for (int i = 0; i < 8; ++i) {
    int d = dBase + i;
    float xv = (v[i] - mean) * rstd * ln_g[d] + ln_b[d];
    xn[(size_t)t * DD + d] = f32_to_bf16u(xv);
  }
}

// =====================================================================
// Kernel 2: counts -> 64-aligned segment offsets, cursors, m-tile map, perm=-1
// =====================================================================
__global__ __launch_bounds__(256) void setup_kernel(
    const int* __restrict__ expert, int* __restrict__ perm,
    int* __restrict__ cursor, int* __restrict__ mtile_expert) {
  __shared__ int cnt[EE];
  __shared__ int off[EE + 1];
  int tid = threadIdx.x;
  if (tid < EE) cnt[tid] = 0;
  __syncthreads();
  for (int t = tid; t < T_TOK; t += 256) atomicAdd(&cnt[expert[t]], 1);
  __syncthreads();
  if (tid == 0) {
    off[0] = 0;
    for (int e = 0; e < EE; ++e) off[e + 1] = off[e] + ((cnt[e] + TILE_M - 1) & ~(TILE_M - 1));
  }
  __syncthreads();
  if (tid < EE) cursor[tid] = off[tid];
  for (int i = tid; i < TPAD; i += 256) perm[i] = -1;
  for (int m = tid; m < NM; m += 256) {
    int r = m * TILE_M;
    int em = -1;
    for (int e = 0; e < EE; ++e)
      if (r >= off[e] && r < off[e + 1]) em = e;
    mtile_expert[m] = em;
  }
}

// =====================================================================
// Kernel 3: scatter token ids into expert-sorted perm
// =====================================================================
__global__ __launch_bounds__(256) void scatter_kernel(
    const int* __restrict__ expert, int* __restrict__ cursor, int* __restrict__ perm) {
  int t = blockIdx.x * 256 + threadIdx.x;
  if (t >= T_TOK) return;
  int pos = atomicAdd(&cursor[expert[t]], 1);
  perm[pos] = t;
}

// =====================================================================
// Grouped GEMM core: tile 64(M) x 128(N), K-step 32, 8 waves.
// Fragment-ordered LDS tiles; each lane fragment = 32 contiguous bytes.
//   Af[(mSub*32+lane)*16 + idx]  idx = (k&15), lane = (m&15)+16*(k>>4)
//   Bf[(n*2+half)*16 + (k&15)]   half: lanes 0-15 -> K 0..15, lanes 16-31 -> K 16..31
// =====================================================================

// ---- Kernel 4: h = relu(xn @ W1[e] + b1[e]) ----
__global__ __launch_bounds__(256) void gemm1_kernel(
    const unsigned short* __restrict__ xn, const float* __restrict__ W1,
    const float* __restrict__ b1, const int* __restrict__ perm,
    const int* __restrict__ mtile_expert, unsigned short* __restrict__ h) {
  __shared__ alignas(32) unsigned short Af[2][4 * 32 * 16];   // 4KB per buf
  __shared__ alignas(32) unsigned short Bf[2][128 * 2 * 16];  // 8KB per buf
  __shared__ int s_rows[TILE_M];

  int m = blockIdx.x;
  int e = mtile_expert[m];
  if (e < 0) return;
  int nTile = blockIdx.y * 128;
  int tid = threadIdx.x;
  if (tid < TILE_M) s_rows[tid] = perm[m * TILE_M + tid];
  __syncthreads();

  int wave = tid >> 5, lane = tid & 31;
  int mIdx = wave & 3;     // m-subtile (16 rows)
  int nGrp = wave >> 2;    // 64-col group
  v8f c[4] = {};

  // A staging: thread -> (row, 8 consecutive k). Pad rows (<0) gather token 0;
  // their results are discarded at the gated scatter, so no masking needed.
  int aRow = tid >> 2;
  int aCol = (tid & 3) * 8;
  int aL   = (aRow & 15) + ((aCol >> 4) << 4);
  int aDst = ((aRow >> 4) * 32 + aL) * 16 + ((aCol >> 3) & 1) * 8;
  int prow = s_rows[aRow];
  if (prow < 0) prow = 0;
  const uint4* aSrc = (const uint4*)(xn + (size_t)prow * DD + aCol);

  // B staging: thread -> (one column n, 16 consecutive k in its half)
  int bN = tid & 127;
  int bHalf = tid >> 7;
  const float* bSrc = W1 + (size_t)e * DD * FF + nTile + bN;
  int bDst = (bN * 2 + bHalf) * 16;

  auto stage = [&](int buf, int k0) {
    uint4 av = aSrc[k0 >> 3];
    *(uint4*)&Af[buf][aDst] = av;
    const float* wp = bSrc + (size_t)(k0 + bHalf * 16) * FF;
    if (k0 + 64 < DD) __builtin_prefetch(wp + (size_t)32 * FF, 0, 1);
    float f[16];
#pragma unroll
    for (int i = 0; i < 16; ++i) f[i] = wp[(size_t)i * FF];
    unsigned int u[8];
#pragma unroll
    for (int i = 0; i < 8; ++i) u[i] = pack_bf16_pair(f[2 * i], f[2 * i + 1]);
    *(uint4*)&Bf[buf][bDst]     = make_uint4(u[0], u[1], u[2], u[3]);
    *(uint4*)&Bf[buf][bDst + 8] = make_uint4(u[4], u[5], u[6], u[7]);
  };

  auto compute = [&](int buf) {
    // issue all fragment loads first, then WMMAs back-to-back
    const unsigned short* afp = &Af[buf][(mIdx * 32 + lane) * 16];
    const unsigned short* bfp = &Bf[buf][((nGrp * 64 + (lane & 15)) * 2 + (lane >> 4)) * 16];
    v16bf a  = *(const v16bf*)afp;
    v16bf b0 = *(const v16bf*)(bfp);
    v16bf b1 = *(const v16bf*)(bfp + 16 * 32);
    v16bf b2 = *(const v16bf*)(bfp + 32 * 32);
    v16bf b3 = *(const v16bf*)(bfp + 48 * 32);
    c[0] = __builtin_amdgcn_wmma_f32_16x16x32_bf16(false, a, false, b0, (short)0, c[0], false, false);
    c[1] = __builtin_amdgcn_wmma_f32_16x16x32_bf16(false, a, false, b1, (short)0, c[1], false, false);
    c[2] = __builtin_amdgcn_wmma_f32_16x16x32_bf16(false, a, false, b2, (short)0, c[2], false, false);
    c[3] = __builtin_amdgcn_wmma_f32_16x16x32_bf16(false, a, false, b3, (short)0, c[3], false, false);
  };

  stage(0, 0);
#pragma unroll 2
  for (int kI = 0; kI < DD / 32; ++kI) {
    int p = kI & 1;
    __syncthreads();
    if (kI + 1 < DD / 32) stage(p ^ 1, (kI + 1) * 32);
    compute(p);
  }

  // epilogue: bias + relu -> bf16 h at permuted position
#pragma unroll
  for (int t = 0; t < 4; ++t) {
    int col = nTile + nGrp * 64 + t * 16 + (lane & 15);
    float bias = b1[(size_t)e * FF + col];
#pragma unroll
    for (int r = 0; r < 8; ++r) {
      int row = mIdx * 16 + (lane >> 4) * 8 + r;
      size_t hrow = (size_t)m * TILE_M + row;
      float v = fmaxf(c[t][r] + bias, 0.0f);
      h[hrow * FF + col] = f32_to_bf16u(v);
    }
  }
}

// ---- Kernel 5: y = (h @ W2[e] + b2[e]) * gate, scattered to out ----
__global__ __launch_bounds__(256) void gemm2_kernel(
    const unsigned short* __restrict__ h, const float* __restrict__ W2,
    const float* __restrict__ b2, const int* __restrict__ perm,
    const int* __restrict__ mtile_expert, const float* __restrict__ gate,
    float* __restrict__ out) {
  __shared__ alignas(32) unsigned short Af[2][4 * 32 * 16];
  __shared__ alignas(32) unsigned short Bf[2][128 * 2 * 16];
  __shared__ int s_rows[TILE_M];
  __shared__ float s_gate[TILE_M];

  int m = blockIdx.x;
  int e = mtile_expert[m];
  if (e < 0) return;
  int nTile = blockIdx.y * 128;
  int tid = threadIdx.x;
  if (tid < TILE_M) {
    int pr = perm[m * TILE_M + tid];
    s_rows[tid] = pr;
    s_gate[tid] = (pr >= 0) ? gate[pr] : 0.0f;
  }
  __syncthreads();

  int wave = tid >> 5, lane = tid & 31;
  int mIdx = wave & 3;
  int nGrp = wave >> 2;
  v8f c[4] = {};

  int aRow = tid >> 2;
  int aCol = (tid & 3) * 8;
  int aL   = (aRow & 15) + ((aCol >> 4) << 4);
  int aDst = ((aRow >> 4) * 32 + aL) * 16 + ((aCol >> 3) & 1) * 8;
  const uint4* aSrc = (const uint4*)(h + ((size_t)m * TILE_M + aRow) * FF + aCol);

  int bN = tid & 127;
  int bHalf = tid >> 7;
  const float* bSrc = W2 + (size_t)e * FF * DD + nTile + bN;
  int bDst = (bN * 2 + bHalf) * 16;

  auto stage = [&](int buf, int k0) {
    uint4 av = aSrc[k0 >> 3];
    *(uint4*)&Af[buf][aDst] = av;
    const float* wp = bSrc + (size_t)(k0 + bHalf * 16) * DD;
    if (k0 + 64 < FF) __builtin_prefetch(wp + (size_t)32 * DD, 0, 1);
    float f[16];
#pragma unroll
    for (int i = 0; i < 16; ++i) f[i] = wp[(size_t)i * DD];
    unsigned int u[8];
#pragma unroll
    for (int i = 0; i < 8; ++i) u[i] = pack_bf16_pair(f[2 * i], f[2 * i + 1]);
    *(uint4*)&Bf[buf][bDst]     = make_uint4(u[0], u[1], u[2], u[3]);
    *(uint4*)&Bf[buf][bDst + 8] = make_uint4(u[4], u[5], u[6], u[7]);
  };

  auto compute = [&](int buf) {
    const unsigned short* afp = &Af[buf][(mIdx * 32 + lane) * 16];
    const unsigned short* bfp = &Bf[buf][((nGrp * 64 + (lane & 15)) * 2 + (lane >> 4)) * 16];
    v16bf a  = *(const v16bf*)afp;
    v16bf b0 = *(const v16bf*)(bfp);
    v16bf b1 = *(const v16bf*)(bfp + 16 * 32);
    v16bf b2 = *(const v16bf*)(bfp + 32 * 32);
    v16bf b3 = *(const v16bf*)(bfp + 48 * 32);
    c[0] = __builtin_amdgcn_wmma_f32_16x16x32_bf16(false, a, false, b0, (short)0, c[0], false, false);
    c[1] = __builtin_amdgcn_wmma_f32_16x16x32_bf16(false, a, false, b1, (short)0, c[1], false, false);
    c[2] = __builtin_amdgcn_wmma_f32_16x16x32_bf16(false, a, false, b2, (short)0, c[2], false, false);
    c[3] = __builtin_amdgcn_wmma_f32_16x16x32_bf16(false, a, false, b3, (short)0, c[3], false, false);
  };

  stage(0, 0);
#pragma unroll 2
  for (int kI = 0; kI < FF / 32; ++kI) {
    int p = kI & 1;
    __syncthreads();
    if (kI + 1 < FF / 32) stage(p ^ 1, (kI + 1) * 32);
    compute(p);
  }

#pragma unroll
  for (int t = 0; t < 4; ++t) {
    int col = nTile + nGrp * 64 + t * 16 + (lane & 15);
    float bias = b2[(size_t)e * DD + col];
#pragma unroll
    for (int r = 0; r < 8; ++r) {
      int row = mIdx * 16 + (lane >> 4) * 8 + r;
      int tok = s_rows[row];
      if (tok >= 0) {
        out[(size_t)tok * DD + col] = (c[t][r] + bias) * s_gate[row];
      }
    }
  }
}

// =====================================================================
// Host launcher
// =====================================================================
static inline size_t align256(size_t x) { return (x + 255) & ~(size_t)255; }

extern "C" void kernel_launch(void* const* d_in, const int* in_sizes, int n_in,
                              void* d_out, int out_size, void* d_ws, size_t ws_size,
                              hipStream_t stream) {
  const float* x    = (const float*)d_in[0];
  const float* Wr   = (const float*)d_in[1];
  const float* br   = (const float*)d_in[2];
  const float* ln_g = (const float*)d_in[3];
  const float* ln_b = (const float*)d_in[4];
  const float* W1   = (const float*)d_in[5];
  const float* b1   = (const float*)d_in[6];
  const float* W2   = (const float*)d_in[7];
  const float* b2   = (const float*)d_in[8];
  float* out = (float*)d_out;

  char* ws = (char*)d_ws;
  size_t off = 0;
  unsigned short* xn_ws = (unsigned short*)(ws + off); off = align256(off + (size_t)T_TOK * DD * 2);
  unsigned short* h_ws  = (unsigned short*)(ws + off); off = align256(off + (size_t)TPAD * FF * 2);
  float* gate_ws        = (float*)(ws + off);          off = align256(off + (size_t)T_TOK * 4);
  int* expert_ws        = (int*)(ws + off);            off = align256(off + (size_t)T_TOK * 4);
  int* perm_ws          = (int*)(ws + off);            off = align256(off + (size_t)TPAD * 4);
  int* cursor_ws        = (int*)(ws + off);            off = align256(off + (size_t)EE * 4);
  int* mtile_ws         = (int*)(ws + off);            off = align256(off + (size_t)NM * 4);

  router_ln_kernel<<<T_TOK, 128, 0, stream>>>(x, Wr, br, ln_g, ln_b, xn_ws, gate_ws, expert_ws);
  setup_kernel<<<1, 256, 0, stream>>>(expert_ws, perm_ws, cursor_ws, mtile_ws);
  scatter_kernel<<<(T_TOK + 255) / 256, 256, 0, stream>>>(expert_ws, cursor_ws, perm_ws);
  gemm1_kernel<<<dim3(NM, FF / 128), 256, 0, stream>>>(xn_ws, W1, b1, perm_ws, mtile_ws, h_ws);
  gemm2_kernel<<<dim3(NM, DD / 128), 256, 0, stream>>>(h_ws, W2, b2, perm_ws, mtile_ws, gate_ws, out);
}
